// CHEB_42992622633741
// MI455X (gfx1250) — compile-verified, matrix operand order checked by reading, not run on hardware
//
#include <hip/hip_runtime.h>
#include <hip/hip_bf16.h>

// ---------- types ----------
typedef __attribute__((ext_vector_type(16))) __bf16        v16bf;
typedef __attribute__((ext_vector_type(8)))  float         v8f;
typedef __attribute__((ext_vector_type(8)))  unsigned int  v8u;

union Frag { v8u u; v16bf bf; };

#define GNN_N   100000
#define GNN_E   1600000
#define GNN_HID 128
#define GNN_OUT 64

__device__ __forceinline__ unsigned short f2bf(float f) {
    unsigned int u = __float_as_uint(f);
    unsigned int r = u + 0x7FFFu + ((u >> 16) & 1u);   // round-to-nearest-even
    return (unsigned short)(r >> 16);
}

__device__ __forceinline__ v16bf mkfrag(uint4 lo, uint4 hi) {
    Frag f;
    f.u[0] = lo.x; f.u[1] = lo.y; f.u[2] = lo.z; f.u[3] = lo.w;
    f.u[4] = hi.x; f.u[5] = hi.y; f.u[6] = hi.z; f.u[7] = hi.w;
    return f.bf;
}

// ---------- utility: zero fill (graph-capture safe, no hipMemset) ----------
__global__ void k_zero(float* __restrict__ p, long n) {
    long i = (long)blockIdx.x * blockDim.x + threadIdx.x;
    long stride = (long)gridDim.x * blockDim.x;
    for (; i < n; i += stride) p[i] = 0.0f;
}

// ---------- degrees ----------
__global__ void k_deg(const int* __restrict__ src, const int* __restrict__ dst,
                      float* __restrict__ deg_in, float* __restrict__ deg_out, int E) {
    int e = blockIdx.x * blockDim.x + threadIdx.x;
    if (e < E) {
        atomicAdd(&deg_in[dst[e]], 1.0f);
        atomicAdd(&deg_out[src[e]], 1.0f);
    }
}

__global__ void k_norms(const float* __restrict__ deg_in, const float* __restrict__ deg_out,
                        float* __restrict__ norm_in, float* __restrict__ norm_out,
                        float* __restrict__ inv_deg_in, int n) {
    int i = blockIdx.x * blockDim.x + threadIdx.x;
    if (i < n) {
        float di = fmaxf(deg_in[i], 1.0f);
        float dg = fmaxf(deg_out[i], 1.0f);
        norm_in[i]    = rsqrtf(di);
        norm_out[i]   = rsqrtf(dg);
        inv_deg_in[i] = 1.0f / di;
    }
}

// ---------- edge scatter: y[dst] += x[src] * (sscale ? sscale[src] : 1) ----------
// one wave32 per edge, 4 columns per lane (128 columns)
__global__ void k_scatter(const float* __restrict__ x,
                          const int* __restrict__ src, const int* __restrict__ dst,
                          const float* __restrict__ sscale,
                          float* __restrict__ y, int E) {
    int t = blockIdx.x * blockDim.x + threadIdx.x;
    int e = t >> 5;
    int lane = t & 31;
    if (e >= E) return;
    int s = src[e];
    int d = dst[e];
    float sc = sscale ? sscale[s] : 1.0f;
    const float* xr = x + (long)s * GNN_HID;
    float*       yr = y + (long)d * GNN_HID;
#pragma unroll
    for (int c = 0; c < 4; ++c) {
        int col = lane + c * 32;
        atomicAdd(&yr[col], xr[col] * sc);
    }
}

// ---------- row-scaled bf16 conversion: ob[i] = bf16(x[i] * rs[row]) ----------
__global__ void k_rowscale_bf16(const float* __restrict__ x, const float* __restrict__ rs,
                                unsigned short* __restrict__ ob, long n) {
    long i = (long)blockIdx.x * blockDim.x + threadIdx.x;
    if (i < n) {
        long row = i >> 7;   // 128 cols
        ob[i] = f2bf(x[i] * rs[row]);
    }
}

// ---------- X1 = -Ahat(X0): x1 = -t * norm_in[row]  (f32 + bf16 outputs) ----------
__global__ void k_x1(const float* __restrict__ t, const float* __restrict__ norm_in,
                     float* __restrict__ x1f, unsigned short* __restrict__ x1b, long n) {
    long i = (long)blockIdx.x * blockDim.x + threadIdx.x;
    if (i < n) {
        long row = i >> 7;
        float v = -t[i] * norm_in[row];
        x1f[i] = v;
        x1b[i] = f2bf(v);
    }
}

// ---------- X2 = -2*Ahat(X1) - X0 ----------
__global__ void k_x2(const float* __restrict__ t, const float* __restrict__ norm_in,
                     const float* __restrict__ x0f, unsigned short* __restrict__ x2b, long n) {
    long i = (long)blockIdx.x * blockDim.x + threadIdx.x;
    if (i < n) {
        long row = i >> 7;
        float v = -2.0f * t[i] * norm_in[row] - x0f[i];
        x2b[i] = f2bf(v);
    }
}

// ---------- weight pack: f32 KxNcols (K=128) -> bf16-pair dwords in WMMA B layout ----------
// layout: [nt][kt (4)][lane (32)][vgpr (8)] ; word = { W[k][n], W[k+1][n] }
// lane L: n = nt*16 + (L&15); k = kt*32 + ((v&4)<<2) + (L>>4)*8 + ((v&3)<<1)
__global__ void k_pack_w(const float* __restrict__ W, unsigned int* __restrict__ out,
                         int ncols, int nwords) {
    int idx = blockIdx.x * blockDim.x + threadIdx.x;
    if (idx >= nwords) return;
    int v    = idx & 7;
    int lane = (idx >> 3) & 31;
    int kt   = (idx >> 8) & 3;
    int nt   = idx >> 10;
    int n    = nt * 16 + (lane & 15);
    int half = lane >> 4;
    int k    = kt * 32 + ((v & 4) << 2) + half * 8 + ((v & 3) << 1);
    unsigned int lo = f2bf(W[(long)k       * ncols + n]);
    unsigned int hi = f2bf(W[(long)(k + 1) * ncols + n]);
    out[idx] = lo | (hi << 16);
}

// ---------- WMMA GEMM: out = [relu]( A(Mx128,bf16) @ Bp + bias + addend ) ----------
// NT = number of 16-col tiles (8 -> 128 cols, 4 -> 64 cols). K fixed at 128 (4 k-tiles).
// One wave per 16-row strip; full EXEC within active waves (N % 16 == 0).
template <int NT>
__global__ void k_gemm(const unsigned short* __restrict__ A,
                       const unsigned int* __restrict__ Bp,
                       const float* __restrict__ bias,
                       const float* __restrict__ addend,
                       float* __restrict__ outF,
                       unsigned short* __restrict__ outB,
                       int relu, int nTiles) {
    const int NC = NT * 16;
    int wave = (int)((blockIdx.x * blockDim.x + threadIdx.x) >> 5);
    int lane = threadIdx.x & 31;
    if (wave >= nTiles) return;

    int row0 = wave * 16;
    int half = lane >> 4;
    int mA   = row0 + (lane & 15);

    v8f acc[NT] = {};

#pragma unroll
    for (int kt = 0; kt < 4; ++kt) {
        // A fragment: 16 bf16 (8 dwords) per lane, two contiguous b128 loads
        const uint4* ap = (const uint4*)(A + ((long)mA * GNN_HID + kt * 32 + half * 8));
        uint4 alo = ap[0];
        uint4 ahi = ap[2];            // +16 bf16 elements = +32 bytes
        v16bf afrag = mkfrag(alo, ahi);
#pragma unroll
        for (int nt = 0; nt < NT; ++nt) {
            const uint4* bp = (const uint4*)(Bp + (((nt * 4 + kt) * 32 + lane) * 8));
            uint4 blo = bp[0];
            uint4 bhi = bp[1];
            v16bf bfrag = mkfrag(blo, bhi);
            acc[nt] = __builtin_amdgcn_wmma_f32_16x16x32_bf16(
                false, afrag, false, bfrag, (short)0, acc[nt], false, false);
        }
    }

    // epilogue: C layout -> element (m, n): n = lane&15 (per n-tile), m = row0 + half*8 + r
    int nlo  = lane & 15;
    int mrow = row0 + half * 8;
#pragma unroll
    for (int nt = 0; nt < NT; ++nt) {
        int n = nt * 16 + nlo;
        float bval = bias ? bias[n] : 0.0f;
#pragma unroll
        for (int r = 0; r < 8; ++r) {
            long o = (long)(mrow + r) * NC + n;
            float v = acc[nt][r] + bval;
            if (addend) v += addend[o];
            if (relu)   v = fmaxf(v, 0.0f);
            if (outF) outF[o] = v;
            if (outB) outB[o] = f2bf(v);
        }
    }
}

// ---------- host orchestration ----------
extern "C" void kernel_launch(void* const* d_in, const int* in_sizes, int n_in,
                              void* d_out, int out_size, void* d_ws, size_t ws_size,
                              hipStream_t stream) {
    (void)in_sizes; (void)n_in; (void)out_size; (void)ws_size;

    const float* feat    = (const float*)d_in[0];
    const int*   src     = (const int*)d_in[1];
    const int*   dst     = (const int*)d_in[2];
    const float* W1      = (const float*)d_in[3];
    const float* b1      = (const float*)d_in[4];
    const float* W_self  = (const float*)d_in[5];
    const float* W_neigh = (const float*)d_in[6];
    const float* b2      = (const float*)d_in[7];
    const float* W_cheb  = (const float*)d_in[8];
    const float* b3      = (const float*)d_in[9];
    float*       out     = (float*)d_out;

    const long N = GNN_N;
    const int  E = GNN_E;
    const long NH = N * GNN_HID;

    // workspace carve (256B aligned slices)
    char*  ws  = (char*)d_ws;
    size_t off = 0;
    auto carve = [&](size_t bytes) -> void* {
        void* p = ws + off;
        off = (off + bytes + 255) & ~(size_t)255;
        return p;
    };

    float* deg_in     = (float*)carve(N * 4);
    float* deg_out    = (float*)carve(N * 4);
    float* norm_in    = (float*)carve(N * 4);
    float* norm_out   = (float*)carve(N * 4);
    float* inv_deg_in = (float*)carve(N * 4);
    float* agg        = (float*)carve(NH * 4);   // scatter target / GEMM tmp (reused)
    float* h1f        = (float*)carve(NH * 4);
    float* x0f        = (float*)carve(NH * 4);
    float* x1f        = (float*)carve(NH * 4);
    unsigned short* a0b = (unsigned short*)carve(NH * 2);
    unsigned short* h1b = (unsigned short*)carve(NH * 2);
    unsigned short* hnb = (unsigned short*)carve(NH * 2);
    unsigned short* x0b = (unsigned short*)carve(NH * 2);
    unsigned short* x1b = (unsigned short*)carve(NH * 2);
    unsigned short* x2b = (unsigned short*)carve(NH * 2);
    unsigned int* W1p  = (unsigned int*)carve(8192 * 4);
    unsigned int* Wsp  = (unsigned int*)carve(8192 * 4);
    unsigned int* Wnp  = (unsigned int*)carve(8192 * 4);
    unsigned int* Wc0p = (unsigned int*)carve(4096 * 4);
    unsigned int* Wc1p = (unsigned int*)carve(4096 * 4);
    unsigned int* Wc2p = (unsigned int*)carve(4096 * 4);

    const int TB = 256;
    int degBlocks  = (E + TB - 1) / TB;
    int nodeBlocks = (int)((N + TB - 1) / TB);
    int mapBlocks  = (int)((NH + TB - 1) / TB);
    int scatBlocks = (int)(((long)E * 32 + TB - 1) / TB);
    int nTiles     = (int)(N / 16);                       // 6250 exact
    int gemmBlocks = (nTiles * 32 + TB - 1) / TB;
    int zeroBlocks = 2048;

    // degrees + norms
    k_zero<<<zeroBlocks, TB, 0, stream>>>(deg_in, N);
    k_zero<<<zeroBlocks, TB, 0, stream>>>(deg_out, N);
    k_deg<<<degBlocks, TB, 0, stream>>>(src, dst, deg_in, deg_out, E);
    k_norms<<<nodeBlocks, TB, 0, stream>>>(deg_in, deg_out, norm_in, norm_out, inv_deg_in, (int)N);

    // pack weights into WMMA B-fragment layout
    k_pack_w<<<8192 / TB, TB, 0, stream>>>(W1,     W1p, 128, 8192);
    k_pack_w<<<8192 / TB, TB, 0, stream>>>(W_self, Wsp, 128, 8192);
    k_pack_w<<<8192 / TB, TB, 0, stream>>>(W_neigh, Wnp, 128, 8192);
    k_pack_w<<<4096 / TB, TB, 0, stream>>>(W_cheb,                Wc0p, 64, 4096);
    k_pack_w<<<4096 / TB, TB, 0, stream>>>(W_cheb + 128 * GNN_OUT, Wc1p, 64, 4096);
    k_pack_w<<<4096 / TB, TB, 0, stream>>>(W_cheb + 256 * GNN_OUT, Wc2p, 64, 4096);

    // h0 = agg(feat * norm_out[src]) * norm_in  -> bf16
    k_zero<<<zeroBlocks, TB, 0, stream>>>(agg, NH);
    k_scatter<<<scatBlocks, TB, 0, stream>>>(feat, src, dst, norm_out, agg, E);
    k_rowscale_bf16<<<mapBlocks, TB, 0, stream>>>(agg, norm_in, a0b, NH);

    // h1 = relu(h0 @ W1 + b1)  (f32 for scatter, bf16 for next GEMM)
    k_gemm<8><<<gemmBlocks, TB, 0, stream>>>(a0b, W1p, b1, nullptr, h1f, h1b, 1, nTiles);

    // h_neigh = agg(h1) / max(deg_in,1) -> bf16
    k_zero<<<zeroBlocks, TB, 0, stream>>>(agg, NH);
    k_scatter<<<scatBlocks, TB, 0, stream>>>(h1f, src, dst, nullptr, agg, E);
    k_rowscale_bf16<<<mapBlocks, TB, 0, stream>>>(agg, inv_deg_in, hnb, NH);

    // x0 = relu(h1 @ W_self + b2 + h_neigh @ W_neigh)
    k_gemm<8><<<gemmBlocks, TB, 0, stream>>>(h1b, Wsp, b2, nullptr, agg, nullptr, 0, nTiles);
    k_gemm<8><<<gemmBlocks, TB, 0, stream>>>(hnb, Wnp, nullptr, agg, x0f, x0b, 1, nTiles);

    // x1 = -Ahat(x0) ; Ahat(x) = agg(x * norm_in[src]) * norm_in
    k_zero<<<zeroBlocks, TB, 0, stream>>>(agg, NH);
    k_scatter<<<scatBlocks, TB, 0, stream>>>(x0f, src, dst, norm_in, agg, E);
    k_x1<<<mapBlocks, TB, 0, stream>>>(agg, norm_in, x1f, x1b, NH);

    // x2 = -2*Ahat(x1) - x0
    k_zero<<<zeroBlocks, TB, 0, stream>>>(agg, NH);
    k_scatter<<<scatBlocks, TB, 0, stream>>>(x1f, src, dst, norm_in, agg, E);
    k_x2<<<mapBlocks, TB, 0, stream>>>(agg, norm_in, x0f, x2b, NH);

    // out = x0@Wc0 + x1@Wc1 + x2@Wc2 + b3   (K=384 split into 3 accumulating GEMMs)
    k_gemm<4><<<gemmBlocks, TB, 0, stream>>>(x0b, Wc0p, b3, nullptr, out, nullptr, 0, nTiles);
    k_gemm<4><<<gemmBlocks, TB, 0, stream>>>(x1b, Wc1p, nullptr, out, out, nullptr, 0, nTiles);
    k_gemm<4><<<gemmBlocks, TB, 0, stream>>>(x2b, Wc2p, nullptr, out, out, nullptr, 0, nTiles);
}
